// DecoupledContrastiveLoss_16965120819307
// MI455X (gfx1250) — compile-verified, matrix operand order checked by reading, not run on hardware
//
#include <hip/hip_runtime.h>
#include <hip/hip_bf16.h>

typedef __attribute__((ext_vector_type(16))) __bf16        bf16x16;
typedef __attribute__((ext_vector_type(8)))  float         f32x8;
typedef __attribute__((ext_vector_type(8)))  unsigned int  u32x8;
typedef __attribute__((ext_vector_type(4)))  unsigned int  u32x4;
typedef __attribute__((ext_vector_type(8)))  int           i32x8;
typedef __attribute__((ext_vector_type(4)))  int           i32x4;

#define TEMP_INV 14.2857142857f   // 1/0.07
#define NGROUP   2048

#if defined(__has_builtin)
# if __has_builtin(__builtin_amdgcn_tensor_load_to_lds)
#  define DCL_HAS_TDM 1
# else
#  define DCL_HAS_TDM 0
# endif
# if __has_builtin(__builtin_amdgcn_s_wait_tensorcnt)
#  define DCL_WAIT_TENSORCNT0() __builtin_amdgcn_s_wait_tensorcnt(0)
# else
#  define DCL_WAIT_TENSORCNT0() asm volatile("s_wait_tensorcnt 0x0" ::: "memory")
# endif
#else
# define DCL_HAS_TDM 0
# define DCL_WAIT_TENSORCNT0() asm volatile("s_wait_tensorcnt 0x0" ::: "memory")
#endif

// ---------------------------------------------------------------- zero fill
__global__ void dcl_zero_kernel(float* p, long long n) {
    long long i = (long long)blockIdx.x * blockDim.x + threadIdx.x;
    long long stride = (long long)gridDim.x * blockDim.x;
    for (; i < n; i += stride) p[i] = 0.0f;
}

// ------------------------------------------------- normalize -> bf16, group sums
__global__ __launch_bounds__(256)
void dcl_norm_kernel(const float* __restrict__ x, __hip_bfloat16* __restrict__ xn,
                     float* __restrict__ gsum, int* __restrict__ cnt,
                     const int* __restrict__ ids, int D) {
    const int row = blockIdx.x;
    const float* xr = x + (long long)row * D;
    float ss = 0.0f;
    for (int k = threadIdx.x; k < D; k += blockDim.x) { float v = xr[k]; ss += v * v; }
    for (int m = 16; m >= 1; m >>= 1) ss += __shfl_xor(ss, m, 32);
    __shared__ float red[8];
    const int wave = threadIdx.x >> 5, lane = threadIdx.x & 31;
    if (lane == 0) red[wave] = ss;
    __syncthreads();
    float tot = 0.0f;
    for (int i = 0; i < 8; ++i) tot += red[i];
    const float rn = rsqrtf(fmaxf(tot, 1e-20f));
    const int g = ids[row] & (NGROUP - 1);
    if (cnt && threadIdx.x == 0) atomicAdd(&cnt[g], 1);
    for (int k = threadIdx.x; k < D; k += blockDim.x) {
        const float nv = xr[k] * rn;
        xn[(long long)row * D + k] = __float2bfloat16(nv);
        atomicAdd(&gsum[(long long)g * D + k], nv);
    }
}

// ------------------------------------------------- TDM descriptor issue
// Stage one 16-row x 512-col bf16 tile (row stride 512 elems in memory) into LDS
// at byte offset ldsOff, with HW padding of 4 DWORDs after every 256 DWORDs
// (reproduces the +16B/row bank skew -> LDS row stride 1040B).
#if DCL_HAS_TDM
__device__ __forceinline__ void dcl_tdm_stage(const __hip_bfloat16* gsrc, unsigned ldsOff) {
    const unsigned long long ga = (unsigned long long)gsrc;
    u32x4 g0;
    g0[0] = 1u;                                     // count=1 (valid user D#)
    g0[1] = ldsOff;                                 // lds_addr
    g0[2] = (unsigned)ga;                           // global_addr[31:0]
    g0[3] = (unsigned)(ga >> 32) | 0x80000000u;     // global_addr[56:32] | type=2
    i32x8 g1;
    g1[0] = (int)((1u << 16)       // data_size = 2 bytes
                | (1u << 20)       // pad_enable
                | (7u << 22)       // pad_interval: 256 DWORDs (=1024B = one row)
                | (3u << 25));     // pad_amount:   4 DWORDs (=16B skew)
    g1[1] = (int)(512u << 16);     // tensor_dim0 = 512  (bits 79:48)
    g1[2] = (int)(8192u << 16);    // tensor_dim1 = 8192 (bits 111:80)
    g1[3] = (int)(512u << 16);     // tile_dim0 = 512    (bits 127:112)
    g1[4] = 16;                    // tile_dim1 = 16, tile_dim2 = 0
    g1[5] = 512;                   // tensor_dim0_stride = 512
    g1[6] = 0;
    g1[7] = 0;
    const i32x4 z4 = {0, 0, 0, 0};
    const i32x8 z8 = {0, 0, 0, 0, 0, 0, 0, 0};
    // amdgpu-toolchain (clang-23) 6-arg form
    __builtin_amdgcn_tensor_load_to_lds(g0, g1, z4, z4, z8, 0);
}
#endif

// --------------------------------------------- WMMA streaming LSE GEMM
// block: 256 thr (8 waves). Block owns 128 rows x 1024-col slab of sim.
// wave w: rows [rowBase + 16w, +16). A frags live in VGPRs for whole slab.
// B tiles double-buffered in LDS; staged by TDM (tensor_load_to_lds) when
// available, overlapping DMA of tile t+1 with the 16-WMMA compute of tile t.
__global__ __launch_bounds__(256)
void dcl_lse_gemm_kernel(const __hip_bfloat16* __restrict__ A,
                         const __hip_bfloat16* __restrict__ Bm,
                         float* __restrict__ expsum, int maskDiag) {
    constexpr int D = 512, KCH = D / 32;         // 16 k-chunks of 32
    constexpr int LDS_STRIDE = (D + 8) * 2;      // bf16 row + 8-elem skew = 1040 B
    constexpr int NTILES = 64;                   // 64 x 16 cols = 1024-col slab
    const int tid  = threadIdx.x;
    const int lane = tid & 31, wave = tid >> 5;
    const int half = lane >> 4, l15 = lane & 15;
    const int waveRow = blockIdx.x * 128 + wave * 16;
    const int colBase = blockIdx.y * 1024;

    // ---- A fragments (ISA 7.12.2 16-bit A 16x32 layout), held in registers
    bf16x16 afrag[KCH];
    {
        const uint4* arow = reinterpret_cast<const uint4*>(A + (long long)(waveRow + l15) * D);
        #pragma unroll
        for (int c = 0; c < KCH; ++c) {
            const uint4 lo = arow[c * 4 + half];        // K = c*32 + half*8 .. +7
            const uint4 hi = arow[c * 4 + half + 2];    // K = c*32 + half*8 + 16 .. +23
            u32x8 u = { lo.x, lo.y, lo.z, lo.w, hi.x, hi.y, hi.z, hi.w };
            afrag[c] = __builtin_bit_cast(bf16x16, u);
        }
    }

    __shared__ __align__(16) unsigned char ldsB[2][16 * LDS_STRIDE];
    float partial[8] = {0, 0, 0, 0, 0, 0, 0, 0};

#if !DCL_HAS_TDM
    auto stage_manual = [&](int bufIdx, int colTile) {
        const uint4* bsrc = reinterpret_cast<const uint4*>(Bm + (long long)colTile * D);
        #pragma unroll
        for (int i = 0; i < 4; ++i) {
            const int chunk = tid + i * 256;
            const int r = chunk >> 6, o = chunk & 63;     // 64 chunks per row
            const uint4 v = bsrc[r * 64 + o];
            *reinterpret_cast<uint4*>(&ldsB[bufIdx][r * LDS_STRIDE + o * 16]) = v;
        }
    };
#endif

    // prologue: stage tile 0 into buffer 0
#if DCL_HAS_TDM
    if (wave == 0)
        dcl_tdm_stage(Bm + (long long)colBase * D,
                      (unsigned)(unsigned long long)&ldsB[0][0]);
#else
    stage_manual(0, colBase);
#endif

    for (int tile = 0; tile < NTILES; ++tile) {
        const int colTile = colBase + tile * 16;
#if DCL_HAS_TDM
        if (wave == 0) DCL_WAIT_TENSORCNT0();   // tile `tile` has landed in LDS
#endif
        __syncthreads();                        // release all consumers
        if (tile + 1 < NTILES) {                // start DMA/copy of next tile
#if DCL_HAS_TDM
            if (wave == 0)
                dcl_tdm_stage(Bm + (long long)(colTile + 16) * D,
                              (unsigned)(unsigned long long)&ldsB[(tile + 1) & 1][0]);
#else
            stage_manual((tile + 1) & 1, colTile + 16);
#endif
        }

        f32x8 acc = {0, 0, 0, 0, 0, 0, 0, 0};
        const unsigned char* brow = &ldsB[tile & 1][l15 * LDS_STRIDE + half * 16];
        #pragma unroll
        for (int c = 0; c < KCH; ++c) {
            const uint4* bp = reinterpret_cast<const uint4*>(brow + c * 64);
            const uint4 lo = bp[0];
            const uint4 hi = bp[2];
            u32x8 u = { lo.x, lo.y, lo.z, lo.w, hi.x, hi.y, hi.z, hi.w };
            bf16x16 bfrag = __builtin_bit_cast(bf16x16, u);
            acc = __builtin_amdgcn_wmma_f32_16x16x32_bf16(
                    false, afrag[c], false, bfrag, (short)0, acc, false, false);
        }

        const int gcol = colTile + l15;
        #pragma unroll
        for (int r = 0; r < 8; ++r) {
            const int grow = waveRow + r + half * 8;
            // streaming LSE with fixed shift C = 1/T:  exp(sim - C), sim = dot/T
            float e = __expf((acc[r] - 1.0f) * TEMP_INV);
            if (maskDiag && (gcol == grow)) e = 0.0f;
            partial[r] += e;
        }
    }

    // reduce across the 16 lanes sharing each row (xor masks stay in-half)
    #pragma unroll
    for (int r = 0; r < 8; ++r) {
        float s = partial[r];
        s += __shfl_xor(s, 1, 32);
        s += __shfl_xor(s, 2, 32);
        s += __shfl_xor(s, 4, 32);
        s += __shfl_xor(s, 8, 32);
        if (l15 == 0) atomicAdd(&expsum[waveRow + r + half * 8], s);
    }
}

// ------------------------------------------------------------- log finisher
__global__ void dcl_log_kernel(const float* __restrict__ expsum, float* __restrict__ lse, int n) {
    const int i = blockIdx.x * blockDim.x + threadIdx.x;
    if (i < n) lse[i] = __logf(fmaxf(expsum[i], 1e-30f)) + TEMP_INV;
}

// --------------------------------------------- per-group dot products + npos
__global__ __launch_bounds__(128)
void dcl_group_dot_kernel(const float* __restrict__ gV, const float* __restrict__ gT,
                          const int* __restrict__ cnt, float* __restrict__ acc, int D) {
    const int g = blockIdx.x;
    float dvt = 0, dvv = 0, dtt = 0;
    for (int k = threadIdx.x; k < D; k += blockDim.x) {
        const float a = gV[(long long)g * D + k], b = gT[(long long)g * D + k];
        dvt += a * b; dvv += a * a; dtt += b * b;
    }
    for (int m = 16; m >= 1; m >>= 1) {
        dvt += __shfl_xor(dvt, m, 32);
        dvv += __shfl_xor(dvv, m, 32);
        dtt += __shfl_xor(dtt, m, 32);
    }
    __shared__ float red[3][4];
    const int wave = threadIdx.x >> 5, lane = threadIdx.x & 31;
    if (lane == 0) { red[0][wave] = dvt; red[1][wave] = dvv; red[2][wave] = dtt; }
    __syncthreads();
    if (threadIdx.x == 0) {
        float s0 = 0, s1 = 0, s2 = 0;
        for (int i = 0; i < 4; ++i) { s0 += red[0][i]; s1 += red[1][i]; s2 += red[2][i]; }
        atomicAdd(&acc[0], s0);
        atomicAdd(&acc[1], s1);
        atomicAdd(&acc[2], s2);
        const float c = (float)cnt[g];
        atomicAdd(&acc[3], c * c);
    }
}

// ---------------------------------------- weighted LSE sums over rows
__global__ __launch_bounds__(256)
void dcl_row_weight_kernel(const float* __restrict__ lse4, const int* __restrict__ ids,
                           const int* __restrict__ cnt, float* __restrict__ acc, int B) {
    const int i = blockIdx.x * blockDim.x + threadIdx.x;
    float a[4] = {0, 0, 0, 0};
    if (i < B) {
        const float gc = (float)cnt[ids[i] & (NGROUP - 1)];
        a[0] = gc * lse4[i];                      // sum g[i]*lse_row
        a[1] = gc * lse4[B + i];                  // sum g[j]*lse_col
        a[2] = (gc - 1.0f) * lse4[2 * B + i];     // sum (g-1)*lse_v
        a[3] = (gc - 1.0f) * lse4[3 * B + i];     // sum (g-1)*lse_t
    }
    __shared__ float red[4][8];
    const int wave = threadIdx.x >> 5, lane = threadIdx.x & 31;
    #pragma unroll
    for (int q = 0; q < 4; ++q) {
        float s = a[q];
        for (int m = 16; m >= 1; m >>= 1) s += __shfl_xor(s, m, 32);
        if (lane == 0) red[q][wave] = s;
    }
    __syncthreads();
    if (threadIdx.x == 0) {
        #pragma unroll
        for (int q = 0; q < 4; ++q) {
            float s = 0;
            for (int w = 0; w < 8; ++w) s += red[q][w];
            atomicAdd(&acc[4 + q], s);
        }
    }
}

// ---------------------------------------------------------------- finalize
__global__ void dcl_finalize_kernel(const float* __restrict__ acc, float* __restrict__ out, int B) {
    const float npos = fmaxf(acc[3] - (float)B, 1.0f);
    const float S_ct = acc[0] * TEMP_INV;                    // sum sim over match
    const float S_v  = (acc[1] - (float)B) * TEMP_INV;       // sum vsim over pos_off
    const float S_t  = (acc[2] - (float)B) * TEMP_INV;
    const float v2t  = (acc[4] - S_ct) / npos;
    const float t2v  = (acc[5] - S_ct) / npos;
    const float vin  = (acc[6] - S_v) / npos;
    const float tin  = (acc[7] - S_t) / npos;
    out[0] = 0.5f * (v2t + t2v) + 0.5f * vin + 0.5f * tin;
}

extern "C" void kernel_launch(void* const* d_in, const int* in_sizes, int n_in,
                              void* d_out, int out_size, void* d_ws, size_t ws_size,
                              hipStream_t stream) {
    const float* vf  = (const float*)d_in[0];
    const float* tf  = (const float*)d_in[1];
    const int*   ids = (const int*)d_in[2];
    const int B = in_sizes[2];            // 8192
    const int D = in_sizes[0] / B;        // 512

    char* ws = (char*)d_ws;
    size_t off = 0;
    __hip_bfloat16* Vn = (__hip_bfloat16*)(ws + off); off += (size_t)B * D * 2;
    __hip_bfloat16* Tn = (__hip_bfloat16*)(ws + off); off += (size_t)B * D * 2;
    float* gV   = (float*)(ws + off);     off += (size_t)NGROUP * D * 4;
    float* gT   = (float*)(ws + off);     off += (size_t)NGROUP * D * 4;
    int*   cnt  = (int*)(ws + off);       off += (size_t)NGROUP * 4;
    float* esum = (float*)(ws + off);     off += (size_t)4 * B * 4;
    float* lse4 = (float*)(ws + off);     off += (size_t)4 * B * 4;
    float* acc  = (float*)(ws + off);     off += 64;

    // zero accumulators (gV..acc is one contiguous region)
    {
        const long long nz = ((char*)acc + 64 - (char*)gV) / 4;
        dcl_zero_kernel<<<1024, 256, 0, stream>>>(gV, nz);
    }
    // normalize + group sums
    dcl_norm_kernel<<<B, 256, 0, stream>>>(vf, Vn, gV, cnt,     ids, D);
    dcl_norm_kernel<<<B, 256, 0, stream>>>(tf, Tn, gT, nullptr, ids, D);

    // four WMMA LSE GEMMs: v.t^T rows, t.v^T rows (= cols), v.v^T, t.t^T
    dim3 grid(B / 128, B / 1024);
    dcl_lse_gemm_kernel<<<grid, 256, 0, stream>>>(Vn, Tn, esum + 0 * B, 0);
    dcl_lse_gemm_kernel<<<grid, 256, 0, stream>>>(Tn, Vn, esum + 1 * B, 0);
    dcl_lse_gemm_kernel<<<grid, 256, 0, stream>>>(Vn, Vn, esum + 2 * B, 1);
    dcl_lse_gemm_kernel<<<grid, 256, 0, stream>>>(Tn, Tn, esum + 3 * B, 1);

    dcl_log_kernel<<<(4 * B + 255) / 256, 256, 0, stream>>>(esum, lse4, 4 * B);
    dcl_group_dot_kernel<<<NGROUP, 128, 0, stream>>>(gV, gT, cnt, acc, D);
    dcl_row_weight_kernel<<<(B + 255) / 256, 256, 0, stream>>>(lse4, ids, cnt, acc, B);
    dcl_finalize_kernel<<<1, 1, 0, stream>>>(acc, (float*)d_out, B);
}
